// StructureEncoder_61220463837364
// MI455X (gfx1250) — matrix operand order repaired and checked by reading, not compiled, not run on hardware
//
#include <hip/hip_runtime.h>
#include <hip/hip_bf16.h>
#include <hip/hip_fp16.h>

// Problem dimensions (fixed by the reference)
#define NN   12288
#define EE   393216
#define INF  512
#define HIDF 256
#define OUTF 128
#define THRESH 0.5f

typedef __attribute__((ext_vector_type(16))) _Float16 v16h;
typedef __attribute__((ext_vector_type(8)))  _Float16 v8h;
typedef __attribute__((ext_vector_type(8)))  float    v8f;

// ---------------------------------------------------------------------------
// f32 -> f16 conversion (flat)
__global__ void cvt_f16_kernel(const float* __restrict__ in,
                               _Float16* __restrict__ out, int n) {
    int i = blockIdx.x * blockDim.x + threadIdx.x;
    if (i < n) out[i] = (_Float16)in[i];
}

// f32 [K,Nc] row-major -> f16 transposed [Nc,K] row-major
__global__ void cvt_f16_transpose_kernel(const float* __restrict__ in,
                                         _Float16* __restrict__ out,
                                         int K, int Nc) {
    int i = blockIdx.x * blockDim.x + threadIdx.x;
    if (i >= K * Nc) return;
    int k = i / Nc;
    int n = i - k * Nc;
    out[(size_t)n * K + k] = (_Float16)in[i];
}

// zero fill
__global__ void zero_f32_kernel(float* __restrict__ p, int n) {
    int i = blockIdx.x * blockDim.x + threadIdx.x;
    if (i < n) p[i] = 0.0f;
}

// deg = 1.0
__global__ void deg_init_kernel(float* __restrict__ deg) {
    int i = blockIdx.x * blockDim.x + threadIdx.x;
    if (i < NN) deg[i] = 1.0f;
}

// deg[dst[e]] += 1
__global__ void deg_accum_kernel(const long long* __restrict__ dst,
                                 float* __restrict__ deg) {
    int e = blockIdx.x * blockDim.x + threadIdx.x;
    if (e < EE) unsafeAtomicAdd(&deg[(int)dst[e]], 1.0f);
}

// dinv = rsqrt(deg)
__global__ void dinv_kernel(const float* __restrict__ deg,
                            float* __restrict__ dinv) {
    int i = blockIdx.x * blockDim.x + threadIdx.x;
    if (i < NN) dinv[i] = rsqrtf(deg[i]);
}

// AGG[dst] += H[src] * dinv[src] * dinv[dst]   (one block per edge, F threads)
__global__ void edge_agg_kernel(const long long* __restrict__ src,
                                const long long* __restrict__ dst,
                                const float* __restrict__ H,
                                const float* __restrict__ dinv,
                                float* __restrict__ AGG, int F) {
    int e = blockIdx.x;
    int f = threadIdx.x;
    int s = (int)src[e];
    int d = (int)dst[e];
    float w = dinv[s] * dinv[d];
    unsafeAtomicAdd(&AGG[(size_t)d * F + f], H[(size_t)s * F + f] * w);
}

// H1h = f16(relu(AGG1 + H*dinv^2 + b1))
__global__ void epilogue1_kernel(const float* __restrict__ AGG1,
                                 const float* __restrict__ H,
                                 const float* __restrict__ dinv,
                                 const float* __restrict__ b1,
                                 _Float16* __restrict__ H1h) {
    int i = blockIdx.x * blockDim.x + threadIdx.x;
    if (i >= NN * HIDF) return;
    int row = i / HIDF;
    int col = i - row * HIDF;
    float di = dinv[row];
    float v = AGG1[i] + H[i] * di * di + b1[col];
    v = fmaxf(v, 0.0f);
    H1h[i] = (_Float16)v;
}

// Z (in place) = AGG2 + Z*dinv^2 + b2
__global__ void epilogue2_kernel(const float* __restrict__ AGG2,
                                 float* __restrict__ Z,
                                 const float* __restrict__ dinv,
                                 const float* __restrict__ b2) {
    int i = blockIdx.x * blockDim.x + threadIdx.x;
    if (i >= NN * OUTF) return;
    int row = i / OUTF;
    int col = i - row * OUTF;
    float di = dinv[row];
    Z[i] = AGG2[i] + Z[i] * di * di + b2[col];
}

// Row normalize: NZh[row] = f16(Z[row] / ||Z[row]||)  (block = 128 thr = 1 row)
__global__ void rownorm_kernel(const float* __restrict__ Z,
                               _Float16* __restrict__ NZh) {
    __shared__ float s[OUTF];
    int row = blockIdx.x;
    int t = threadIdx.x;
    float v = Z[(size_t)row * OUTF + t];
    s[t] = v * v;
    __syncthreads();
    for (int off = OUTF / 2; off > 0; off >>= 1) {
        if (t < off) s[t] += s[t + off];
        __syncthreads();
    }
    float inv = rsqrtf(s[0]);
    NZh[(size_t)row * OUTF + t] = (_Float16)(v * inv);
}

// ---------------------------------------------------------------------------
// Fragment loaders (wave32, v_wmma_f32_16x16x32_f16 layouts).
// A (16x32, M x K): lane m=lane&15; e<8 -> K=e+8*(lane>=16); e>=8 -> K=16+(e-8)+8*(lane>=16)
__device__ inline v16h load_a_frag(const _Float16* __restrict__ base,
                                   size_t rowStride, int row, int kk,
                                   int hh) {
    const _Float16* ap = base + (size_t)row * rowStride + kk + hh * 8;
    v8h lo = *(const v8h*)ap;
    v8h hi = *(const v8h*)(ap + 16);
    v16h a;
#pragma unroll
    for (int e = 0; e < 8; ++e) { a[e] = lo[e]; a[e + 8] = hi[e]; }
    return a;
}

// B (32x16, K x N) with column `col` stored contiguously (length-K vector):
// e<8 -> K = kk + e + 8*(lane>=16); e>=8 -> K = kk + 16 + (e-8) + 8*(lane>=16)
__device__ inline v16h load_bt_frag(const _Float16* __restrict__ base,
                                    size_t colStride, int col, int kk,
                                    int hh) {
    const _Float16* bp = base + (size_t)col * colStride + kk + hh * 8;
    v8h lo = *(const v8h*)bp;
    v8h hi = *(const v8h*)(bp + 16);
    v16h b;
#pragma unroll
    for (int e = 0; e < 8; ++e) { b[e] = lo[e]; b[e + 8] = hi[e]; }
    return b;
}

// ---------------------------------------------------------------------------
// WMMA GEMM: C[M,Nc] = A[M,K] @ Bt^T, A f16 [M,K] row-major,
// Bt f16 [Nc,K] row-major (pre-transposed weights), C f32.
// One 16x16 tile per wave; K multiple of 32; M,Nc multiples of 16.
__global__ void wmma_gemm_f16_kernel(const _Float16* __restrict__ A,
                                     const _Float16* __restrict__ Bt,
                                     float* __restrict__ C,
                                     int M, int K, int Nc) {
    int wave = (blockIdx.x * blockDim.x + threadIdx.x) >> 5;
    int tilesN = Nc >> 4;
    int m0 = (wave / tilesN) << 4;
    int n0 = (wave % tilesN) << 4;
    if (m0 >= M) return;
    int lane = threadIdx.x & 31;
    int hh   = lane >> 4;     // 0: lanes 0-15, 1: lanes 16-31
    int l16  = lane & 15;

    v8f c = {};
    int arow = m0 + l16;
    int bcol = n0 + l16;
    for (int kk = 0; kk < K; kk += 32) {
        v16h a = load_a_frag(A, (size_t)K, arow, kk, hh);
        v16h b = load_bt_frag(Bt, (size_t)K, bcol, kk, hh);
        c = __builtin_amdgcn_wmma_f32_16x16x32_f16(
                false, a, false, b, (short)0, c, false, false);
    }
#pragma unroll
    for (int r = 0; r < 8; ++r) {
        int row = m0 + r + 8 * hh;
        C[(size_t)row * Nc + bcol] = c[r];
    }
}

// ---------------------------------------------------------------------------
// SIM = NZ @ NZ^T (NZ: [NN, OUTF] f16 row-major). Each wave computes a
// 32x64 block: 2 A fragments x 4 B fragments -> 8 accumulators (64 VGPRs),
// 0.75 fragment-loads per WMMA. Writes sim and float mask (v > 0.5).
__global__ void sim_wmma_kernel(const _Float16* __restrict__ NZ,
                                float* __restrict__ sim,
                                float* __restrict__ mask) {
    int wave = (blockIdx.x * blockDim.x + threadIdx.x) >> 5;
    const int stripsN = NN / 64;            // 192 column strips
    int m0 = (wave / stripsN) << 5;         // 32-row block
    int n0 = (wave % stripsN) << 6;         // 64-col block
    if (m0 >= NN) return;
    int lane = threadIdx.x & 31;
    int hh   = lane >> 4;
    int l16  = lane & 15;

    v8f c[2][4] = {};
#pragma unroll
    for (int kk = 0; kk < OUTF; kk += 32) {
        v16h a0 = load_a_frag(NZ, (size_t)OUTF, m0 + l16,      kk, hh);
        v16h a1 = load_a_frag(NZ, (size_t)OUTF, m0 + 16 + l16, kk, hh);
#pragma unroll
        for (int t = 0; t < 4; ++t) {
            // B = NZ^T: column (n0+16t+l16) of B is a contiguous row of NZ
            v16h b = load_bt_frag(NZ, (size_t)OUTF, n0 + t * 16 + l16, kk, hh);
            c[0][t] = __builtin_amdgcn_wmma_f32_16x16x32_f16(
                          false, a0, false, b, (short)0, c[0][t], false, false);
            c[1][t] = __builtin_amdgcn_wmma_f32_16x16x32_f16(
                          false, a1, false, b, (short)0, c[1][t], false, false);
        }
    }

#pragma unroll
    for (int mt = 0; mt < 2; ++mt) {
#pragma unroll
        for (int t = 0; t < 4; ++t) {
            int col = n0 + t * 16 + l16;
#pragma unroll
            for (int r = 0; r < 8; ++r) {
                int row = m0 + mt * 16 + r + 8 * hh;
                float v = c[mt][t][r];
                size_t idx = (size_t)row * NN + col;
                sim[idx]  = v;
                mask[idx] = (v > THRESH) ? 1.0f : 0.0f;
            }
        }
    }
}

// ---------------------------------------------------------------------------
extern "C" void kernel_launch(void* const* d_in, const int* in_sizes, int n_in,
                              void* d_out, int out_size, void* d_ws, size_t ws_size,
                              hipStream_t stream) {
    const float*     x   = (const float*)d_in[0];
    const long long* ei  = (const long long*)d_in[1];   // int64 [2, E]
    const float*     W1  = (const float*)d_in[2];
    const float*     b1  = (const float*)d_in[3];
    const float*     W2  = (const float*)d_in[4];
    const float*     b2  = (const float*)d_in[5];
    const long long* src = ei;
    const long long* dst = ei + EE;

    // workspace carve (256B aligned)
    char* w = (char*)d_ws;
    auto carve = [&](size_t bytes) -> void* {
        void* p = (void*)w;
        w += (bytes + 255) & ~(size_t)255;
        return p;
    };
    _Float16* Xh   = (_Float16*)carve((size_t)NN * INF * 2);
    _Float16* W1t  = (_Float16*)carve((size_t)INF * HIDF * 2);   // [HIDF, INF]
    _Float16* W2t  = (_Float16*)carve((size_t)HIDF * OUTF * 2);  // [OUTF, HIDF]
    float*    H    = (float*)   carve((size_t)NN * HIDF * 4);
    float*    AGG1 = (float*)   carve((size_t)NN * HIDF * 4);
    _Float16* H1h  = (_Float16*)carve((size_t)NN * HIDF * 2);
    float*    Z    = (float*)   carve((size_t)NN * OUTF * 4);
    float*    AGG2 = (float*)   carve((size_t)NN * OUTF * 4);
    _Float16* NZh  = (_Float16*)carve((size_t)NN * OUTF * 2);
    float*    deg  = (float*)   carve((size_t)NN * 4);
    float*    dinv = (float*)   carve((size_t)NN * 4);

    float* sim  = (float*)d_out;
    float* mask = sim + (size_t)NN * NN;

    const int B256 = 256;

    // operand conversion: X flat, weights transposed for contiguous B frags
    cvt_f16_kernel<<<(NN * INF + 255) / 256, B256, 0, stream>>>(x, Xh, NN * INF);
    cvt_f16_transpose_kernel<<<(INF * HIDF + 255) / 256, B256, 0, stream>>>(
        W1, W1t, INF, HIDF);
    cvt_f16_transpose_kernel<<<(HIDF * OUTF + 255) / 256, B256, 0, stream>>>(
        W2, W2t, HIDF, OUTF);

    // degree / dinv (recomputed every call for determinism)
    deg_init_kernel<<<NN / 256, B256, 0, stream>>>(deg);
    deg_accum_kernel<<<EE / 256, B256, 0, stream>>>(dst, deg);
    dinv_kernel<<<NN / 256, B256, 0, stream>>>(deg, dinv);

    // Layer 1: H = Xh @ W1   (12288 tile-waves)
    {
        int tiles = (NN / 16) * (HIDF / 16);
        wmma_gemm_f16_kernel<<<(tiles * 32) / 256, B256, 0, stream>>>(
            Xh, W1t, H, NN, INF, HIDF);
    }
    zero_f32_kernel<<<(NN * HIDF) / 256, B256, 0, stream>>>(AGG1, NN * HIDF);
    edge_agg_kernel<<<EE, HIDF, 0, stream>>>(src, dst, H, dinv, AGG1, HIDF);
    epilogue1_kernel<<<(NN * HIDF) / 256, B256, 0, stream>>>(AGG1, H, dinv, b1, H1h);

    // Layer 2: Z = H1 @ W2  (6144 tile-waves)
    {
        int tiles = (NN / 16) * (OUTF / 16);
        wmma_gemm_f16_kernel<<<(tiles * 32) / 256, B256, 0, stream>>>(
            H1h, W2t, Z, NN, HIDF, OUTF);
    }
    zero_f32_kernel<<<(NN * OUTF) / 256, B256, 0, stream>>>(AGG2, NN * OUTF);
    edge_agg_kernel<<<EE, OUTF, 0, stream>>>(src, dst, Z, dinv, AGG2, OUTF);
    epilogue2_kernel<<<(NN * OUTF) / 256, B256, 0, stream>>>(AGG2, Z, dinv, b2);

    // Row-normalize -> f16
    rownorm_kernel<<<NN, OUTF, 0, stream>>>(Z, NZh);

    // SIM = NZ @ NZ^T : (12288/32) x (12288/64) = 73728 waves, 32x64 each
    {
        int waves = (NN / 32) * (NN / 64);
        sim_wmma_kernel<<<(waves * 32) / 256, B256, 0, stream>>>(NZh, sim, mask);
    }
}
